// ConnectionTransformer_9483287789751
// MI455X (gfx1250) — matrix operand order repaired and checked by reading, not compiled, not run on hardware
//
#include <hip/hip_runtime.h>
#include <hip/hip_bf16.h>

#define D_MODEL   256
#define NUM_SLOTS 128
#define RANK      32
#define STEPS     6
#define VOCAB     32000
#define SEQ       512
#define BATCH     8
#define BL        (BATCH*SEQ)
#define IPART     4

#define WS_PADL (256*33)   // padded fp32 staging: Ws tile 256 rows x (32+1)
#define WT_PADL (32*257)   // padded fp32 staging: Wt tile 32 rows x (256+1)
#define H_PADL  (8*257)    // padded fp32 staging: h tile 8 rows x (256+1)
#define AG_PADL (64*33)    // padded fp32 staging: gemm A tile 64 rows x (32+1)

typedef __attribute__((ext_vector_type(16))) __bf16       v16bf;
typedef __attribute__((ext_vector_type(8)))  float        v8f;
typedef __attribute__((ext_vector_type(4)))  __bf16       bf16x4;
typedef __attribute__((ext_vector_type(4)))  int          v4i;
typedef __attribute__((ext_vector_type(8)))  int          v8i;
typedef __attribute__((ext_vector_type(4)))  unsigned int v4u;

#define Z8 {0.f,0.f,0.f,0.f,0.f,0.f,0.f,0.f}

#if defined(__has_builtin)
#if __has_builtin(__builtin_amdgcn_tensor_load_to_lds)
#define HAVE_TDM 1
#endif
#endif

static __device__ __forceinline__ v8f wmma_bf16(v16bf a, v16bf b, v8f c) {
  // D(16x16,f32) = A(16x32,bf16) x B(32x16,bf16) + C
  return __builtin_amdgcn_wmma_f32_16x16x32_bf16(false, a, false, b, (short)0, c,
                                                 false, false);
}

static __device__ __forceinline__ bf16x4 cvt4(float4 v) {
  bf16x4 r;
  r[0] = (__bf16)v.x; r[1] = (__bf16)v.y; r[2] = (__bf16)v.z; r[3] = (__bf16)v.w;
  return r;
}

#ifdef HAVE_TDM
// TDM: DMA a rows x cols fp32 tile (row pitch = pitch elements) into LDS,
// optionally inserting LDS padding (pad_amount dwords every 2^(pic+1) dwords).
// Descriptor per CDNA5 ISA sec.8: group0 {count=1, lds_addr, global_addr, type=2},
// group1 {data_size/pad, tensor dims, tile dims, dim0 stride}; groups 2/3 zero (2D).
static __device__ __forceinline__ void tdm_load_tile_f32(
    unsigned lds_addr, const float* gaddr, int rows, int cols, int pitch,
    int pic, int pad) {
  unsigned long long ga = (unsigned long long)(uintptr_t)gaddr;
  v4u g0; v8i g1; v4i g2; v4i g3;
  g0[0] = 1u;                                   // count=1, user mode
  g0[1] = lds_addr;                             // lds_addr
  g0[2] = (unsigned)(ga & 0xffffffffu);         // global_addr lo
  g0[3] = (unsigned)((ga >> 32) & 0x01ffffffu) | (2u << 30);  // addr hi | type=2
  unsigned w0 = 2u << 16;                       // data_size=2 (4 bytes)
  if (pad) w0 |= (1u << 20) | ((unsigned)pic << 22) | ((unsigned)(pad-1) << 25);
  g1[0] = (int)w0;
  g1[1] = (cols & 0xffff) << 16;                // tensor_dim0 low16
  g1[2] = (rows & 0xffff) << 16;                // tensor_dim1 low16
  g1[3] = (cols & 0xffff) << 16;                // tile_dim0
  g1[4] = rows & 0xffff;                        // tile_dim1
  g1[5] = pitch;                                // tensor_dim0_stride low32
  g1[6] = 0; g1[7] = 0;
  g2[0] = 0; g2[1] = 0; g2[2] = 0; g2[3] = 0;
  g3[0] = 0; g3[1] = 0; g3[2] = 0; g3[3] = 0;
#if defined(__clang_major__) && (__clang_major__ >= 23)
  v8i g4; g4[0]=0; g4[1]=0; g4[2]=0; g4[3]=0; g4[4]=0; g4[5]=0; g4[6]=0; g4[7]=0;
  __builtin_amdgcn_tensor_load_to_lds(g0, g1, g2, g3, g4, 0);
#else
  __builtin_amdgcn_tensor_load_to_lds(g0, g1, g2, g3, 0);
#endif
}
#endif

// A fragment: 16x32 (MxK) row-major bf16 tile in LDS, row stride lda (elements).
static __device__ __forceinline__ v16bf load_a_frag(const __bf16* a, int lda, int lane) {
  int m  = lane & 15;
  int kh = (lane >> 4) * 16;
  v16bf f;
#pragma unroll
  for (int v = 0; v < 8; ++v) {
    int kb = 2*v + kh;
    f[2*v]   = a[m*lda + kb];
    f[2*v+1] = a[m*lda + kb + 1];
  }
  return f;
}

// A fragment straight from f32 LDS staging (convert during load)
static __device__ __forceinline__ v16bf load_a_frag_f32(const float* a, int lda, int lane) {
  int m  = lane & 15;
  int kh = (lane >> 4) * 16;
  v16bf f;
#pragma unroll
  for (int v = 0; v < 8; ++v) {
    int kb = 2*v + kh;
    f[2*v]   = (__bf16)a[m*lda + kb];
    f[2*v+1] = (__bf16)a[m*lda + kb + 1];
  }
  return f;
}

// A fragment from f32 staging holding only rows 0..7; rows 8..15 read as zero
static __device__ __forceinline__ v16bf load_a_frag_f32z(const float* a, int lda, int lane) {
  int m  = lane & 15;
  int kh = (lane >> 4) * 16;
  bool live = m < 8;
  v16bf f;
#pragma unroll
  for (int v = 0; v < 8; ++v) {
    int kb = 2*v + kh;
    f[2*v]   = (__bf16)(live ? a[m*lda + kb]     : 0.f);
    f[2*v+1] = (__bf16)(live ? a[m*lda + kb + 1] : 0.f);
  }
  return f;
}

// B fragment from bf16 LDS, element (k,n) at b[n*ldb + k]
static __device__ __forceinline__ v16bf load_b_frag_nk(const __bf16* b, int ldb, int lane) {
  int n  = lane & 15;
  int k0 = (lane >> 4) * 16;
  v16bf f;
#pragma unroll
  for (int v = 0; v < 16; ++v) f[v] = b[n*ldb + k0 + v];
  return f;
}

// B fragment from bf16 LDS, element (k,n) at b[k*ldb + n]
static __device__ __forceinline__ v16bf load_b_frag_kn(const __bf16* b, int ldb, int lane) {
  int n  = lane & 15;
  int k0 = (lane >> 4) * 16;
  v16bf f;
#pragma unroll
  for (int v = 0; v < 16; ++v) f[v] = b[(k0 + v)*ldb + n];
  return f;
}

// B fragment straight from f32 LDS staging (convert during load), (k,n) at b[n*ldb+k]
static __device__ __forceinline__ v16bf load_b_frag_nk_f32(const float* b, int ldb, int lane) {
  int n  = lane & 15;
  int k0 = (lane >> 4) * 16;
  v16bf f;
#pragma unroll
  for (int v = 0; v < 16; ++v) f[v] = (__bf16)b[n*ldb + k0 + v];
  return f;
}

// B fragment straight from f32 LDS staging (convert during load), (k,n) at b[k*ldb+n]
static __device__ __forceinline__ v16bf load_b_frag_kn_f32(const float* b, int ldb, int lane) {
  int n  = lane & 15;
  int k0 = (lane >> 4) * 16;
  v16bf f;
#pragma unroll
  for (int v = 0; v < 16; ++v) f[v] = (__bf16)b[(k0 + v)*ldb + n];
  return f;
}

// ---------------------------------------------------------------- embeddings
__global__ void embed_kernel(const int* __restrict__ ids, const float* __restrict__ tok,
                             const float* __restrict__ pos, float* __restrict__ X) {
  unsigned idx = blockIdx.x * 256 + threadIdx.x;  // float4 index over BL*D/4
  unsigned d4 = idx & 63;                         // D/4 = 64
  unsigned t  = idx >> 6;                         // token b*L+l
  unsigned l  = t & (SEQ - 1);
  float4 a = ((const float4*)tok)[(size_t)((unsigned)ids[t]*64u + d4)];
  float4 p = ((const float4*)pos)[l*64u + d4];
  ((float4*)X)[idx] = make_float4(a.x+p.x, a.y+p.y, a.z+p.z, a.w+p.w);
}

// --------------------------------------------- generic C[M,N] = A[M,K] * W[N,K]^T
// Requires M % 64 == 0, N % 128 == 0, K % 32 == 0 (true for every launch here).
// Both A and W tiles streamed by TDM into double-buffered padded f32 staging;
// fragments convert f32->bf16 during load. Waves do no VMEM work in the k-loop.
__global__ __launch_bounds__(256) void gemm_awt(
    const float* __restrict__ Aq, const float* __restrict__ W, float* __restrict__ C,
    int M, int N, int K)
{
  __shared__ __align__(128) float sAst[2][64][33];
  __shared__ __align__(128) float sBst[2][128][33];
  int tid = threadIdx.x, lane = tid & 31, wave = tid >> 5;
  unsigned n0 = blockIdx.x * 128u, m0 = blockIdx.y * 64u;
  v8f acc[4] = {Z8, Z8, Z8, Z8};
  unsigned nk = (unsigned)K >> 5;
#ifdef HAVE_TDM
  if (wave == 0) {                                // prologue DMA pair for kk=0
    tdm_load_tile_f32((unsigned)(uintptr_t)&sAst[0][0][0],
                      Aq + m0*(unsigned)K, 64, 32, K, 4, 1);
    tdm_load_tile_f32((unsigned)(uintptr_t)&sBst[0][0][0],
                      W + n0*(unsigned)K, 128, 32, K, 4, 1);
  }
#endif
  for (unsigned kk = 0; kk < nk; ++kk) {
    unsigned k0 = kk << 5;
    int bi = (int)(kk & 1u);
#ifdef HAVE_TDM
    if (wave == 0 && kk + 1 < nk) {               // DMA next tiles into other buffers
      tdm_load_tile_f32((unsigned)(uintptr_t)&sAst[bi^1][0][0],
                        Aq + (m0*(unsigned)K + k0 + 32), 64, 32, K, 4, 1);
      tdm_load_tile_f32((unsigned)(uintptr_t)&sBst[bi^1][0][0],
                        W + (n0*(unsigned)K + k0 + 32), 128, 32, K, 4, 1);
    }
    if (wave == 0) {
      if (kk + 1 < nk) __builtin_amdgcn_s_wait_tensorcnt(2);  // current pair done
      else             __builtin_amdgcn_s_wait_tensorcnt(0);
    }
#else
#pragma unroll
    for (int it = 0; it < 2; ++it) {              // fallback: 64x32 A tile
      unsigned idx = tid + it*256u, r = idx >> 3, c = (idx & 7)*4;
      float4 v = *(const float4*)(Aq + ((m0 + r)*(unsigned)K + k0 + c));
      float* dst = &sAst[bi][0][0] + r*33u + c;
      dst[0] = v.x; dst[1] = v.y; dst[2] = v.z; dst[3] = v.w;
    }
#pragma unroll
    for (int it = 0; it < 4; ++it) {              // fallback: 128x32 W tile
      unsigned idx = tid + it*256u, r = idx >> 3, c = (idx & 7)*4;
      float4 v = *(const float4*)(W + ((n0 + r)*(unsigned)K + k0 + c));
      float* dst = &sBst[bi][0][0] + r*33u + c;
      dst[0] = v.x; dst[1] = v.y; dst[2] = v.z; dst[3] = v.w;
    }
#endif
    __syncthreads();
    v16bf bfr = load_b_frag_nk_f32(&sBst[bi][wave*16][0], 33, lane);
#pragma unroll
    for (int t = 0; t < 4; ++t) {
      v16bf af = load_a_frag_f32(&sAst[bi][t*16][0], 33, lane);
      acc[t] = wmma_bf16(af, bfr, acc[t]);
    }
    __syncthreads();
  }
  unsigned n = n0 + wave*16u + (lane & 15);
  unsigned mb = (lane >> 4) * 8;
#pragma unroll
  for (int t = 0; t < 4; ++t)
#pragma unroll
    for (int r = 0; r < 8; ++r)
      C[(size_t)((m0 + t*16u + mb + r)*(unsigned)N + n)] = acc[t][r];
}

// ------------------------------------ compress: scores + softmax + mask -> A(B,L,S)
__global__ __launch_bounds__(256) void compress_scores_kernel(
    const float* __restrict__ Q, const float* __restrict__ Ksl,
    const int* __restrict__ amask, float* __restrict__ A)
{
  __shared__ __bf16 sQ[64][36];
  __shared__ __bf16 sK[128][36];
  __shared__ float  sS[64][129];
  int tid = threadIdx.x, lane = tid & 31, wave = tid >> 5;
  unsigned b = blockIdx.y, l0 = blockIdx.x * 64u;
  v8f acc[4] = {Z8, Z8, Z8, Z8};
  for (unsigned k0 = 0; k0 < D_MODEL; k0 += 32) {
    float4 rq[2], rk[4];
#pragma unroll
    for (int it = 0; it < 2; ++it) {
      unsigned idx = tid + it*256u, r = idx >> 3, c4 = idx & 7;
      rq[it] = *(const float4*)(Q + ((b*SEQ + l0 + r)*D_MODEL + k0 + c4*4u));
    }
#pragma unroll
    for (int it = 0; it < 4; ++it) {
      unsigned idx = tid + it*256u, r = idx >> 3, c4 = idx & 7;
      rk[it] = *(const float4*)(Ksl + (r*D_MODEL + k0 + c4*4u));
    }
#pragma unroll
    for (int it = 0; it < 2; ++it) {
      unsigned idx = tid + it*256u, r = idx >> 3, c4 = idx & 7;
      *(bf16x4*)&sQ[r][c4*4] = cvt4(rq[it]);
    }
#pragma unroll
    for (int it = 0; it < 4; ++it) {
      unsigned idx = tid + it*256u, r = idx >> 3, c4 = idx & 7;
      *(bf16x4*)&sK[r][c4*4] = cvt4(rk[it]);
    }
    __syncthreads();
    v16bf bfr = load_b_frag_nk(&sK[wave*16][0], 36, lane);
#pragma unroll
    for (int t = 0; t < 4; ++t) {
      v16bf af = load_a_frag(&sQ[t*16][0], 36, lane);
      acc[t] = wmma_bf16(af, bfr, acc[t]);
    }
    __syncthreads();
  }
  {
    int n = lane & 15, mb = (lane >> 4) * 8;
#pragma unroll
    for (int t = 0; t < 4; ++t)
#pragma unroll
      for (int r = 0; r < 8; ++r)
        sS[t*16 + mb + r][wave*16 + n] = acc[t][r] * 0.0625f;  // 1/sqrt(256)
  }
  __syncthreads();
#pragma unroll
  for (int rr = 0; rr < 8; ++rr) {               // wave handles 8 rows
    unsigned row = wave*8 + rr;
    float mval = (float)amask[b*SEQ + l0 + row];
    float v[4]; float mx = -1e30f;
#pragma unroll
    for (int e = 0; e < 4; ++e) { v[e] = sS[row][lane*4 + e]; mx = fmaxf(mx, v[e]); }
#pragma unroll
    for (int o = 16; o > 0; o >>= 1) mx = fmaxf(mx, __shfl_xor(mx, o));
    float sum = 0.f;
#pragma unroll
    for (int e = 0; e < 4; ++e) { v[e] = __expf(v[e] - mx); sum += v[e]; }
#pragma unroll
    for (int o = 16; o > 0; o >>= 1) sum += __shfl_xor(sum, o);
    float inv = mval / sum;
    *(float4*)&A[((b*SEQ + l0 + row)*NUM_SLOTS + lane*4u)] =
        make_float4(v[0]*inv, v[1]*inv, v[2]*inv, v[3]*inv);
  }
}

// ------------------------------------------------- colsum over L per (b,s)
__global__ void colsum_kernel(const float* __restrict__ A, float* __restrict__ cs) {
  unsigned idx = blockIdx.x * 256 + threadIdx.x;  // B*S threads
  unsigned b = idx >> 7, s = idx & 127;
  float sum = 0.f;
  for (unsigned l = 0; l < SEQ; ++l) sum += A[(b*SEQ + l)*NUM_SLOTS + s];
  cs[idx] = sum + 1e-8f;
}

// --------------------- H_state[b,s,d] = H[s,d] + (1/colsum) * sum_l A[b,l,s] V[b,l,d]
__global__ __launch_bounds__(256) void compress_ir_kernel(
    const float* __restrict__ A, const float* __restrict__ csum,
    const float* __restrict__ Vin, const float* __restrict__ H,
    float* __restrict__ Hst)
{
  __shared__ __bf16 sA[64][36];
  __shared__ __bf16 sB[32][132];
  int tid = threadIdx.x, lane = tid & 31, wave = tid >> 5;
  unsigned n0 = blockIdx.x * 128u, s0 = blockIdx.y * 64u, b = blockIdx.z;
  v8f acc[4] = {Z8, Z8, Z8, Z8};
  for (unsigned k0 = 0; k0 < SEQ; k0 += 32) {
    float4 rat[2], rv[4];
#pragma unroll
    for (int it = 0; it < 2; ++it) {             // A^T tile: float4 along s
      unsigned idx = tid + it*256u, c = idx >> 4, r4 = (idx & 15) * 4;
      rat[it] = *(const float4*)&A[((b*SEQ + k0 + c)*NUM_SLOTS + s0 + r4)];
    }
#pragma unroll
    for (int it = 0; it < 4; ++it) {             // V tile 32x128 (k-major)
      unsigned idx = tid + it*256u, r = idx >> 5, c4 = idx & 31;
      rv[it] = *(const float4*)(Vin + ((b*SEQ + k0 + r)*D_MODEL + n0 + c4*4u));
    }
#pragma unroll
    for (int it = 0; it < 2; ++it) {             // scatter transpose into sA
      unsigned idx = tid + it*256u, c = idx >> 4, r4 = (idx & 15) * 4;
      sA[r4+0][c] = (__bf16)rat[it].x; sA[r4+1][c] = (__bf16)rat[it].y;
      sA[r4+2][c] = (__bf16)rat[it].z; sA[r4+3][c] = (__bf16)rat[it].w;
    }
#pragma unroll
    for (int it = 0; it < 4; ++it) {
      unsigned idx = tid + it*256u, r = idx >> 5, c4 = idx & 31;
      *(bf16x4*)&sB[r][c4*4] = cvt4(rv[it]);
    }
    __syncthreads();
    v16bf bfr = load_b_frag_kn(&sB[0][wave*16], 132, lane);
#pragma unroll
    for (int t = 0; t < 4; ++t) {
      v16bf af = load_a_frag(&sA[t*16][0], 36, lane);
      acc[t] = wmma_bf16(af, bfr, acc[t]);
    }
    __syncthreads();
  }
  unsigned n = lane & 15, mb = (lane >> 4) * 8;
#pragma unroll
  for (int t = 0; t < 4; ++t)
#pragma unroll
    for (int r = 0; r < 8; ++r) {
      unsigned s = s0 + t*16u + mb + r;
      float inv = 1.f / csum[b*NUM_SLOTS + s];
      unsigned d = n0 + wave*16u + n;
      Hst[((b*NUM_SLOTS + s)*D_MODEL + d)] = H[(s*D_MODEL + d)] + acc[t][r] * inv;
    }
}

// ------------------------------------------------- one relational step (partial)
// block (j, p): partial infl over i in [p*32, p*32+32), i != j.
// Ws, Wt and h tiles all streamed by TDM into double-buffered padded f32 staging
// (dynamic LDS); fragments convert f32->bf16 during load.
__global__ __launch_bounds__(256) void step_partial_kernel(
    const float* __restrict__ hin, const float* __restrict__ Wsrc,
    const float* __restrict__ Wtgt, float* __restrict__ Pbuf)
{
  __shared__ __bf16 sP[16][36];
  extern __shared__ float dyn[];                 // 2*(WS_PADL+WT_PADL+H_PADL) floats
  float* stWs0 = dyn;
  float* stWs1 = dyn + WS_PADL;
  float* stWt0 = dyn + 2*WS_PADL;
  float* stWt1 = dyn + 2*WS_PADL + WT_PADL;
  float* stH0  = dyn + 2*WS_PADL + 2*WT_PADL;
  float* stH1  = dyn + 2*WS_PADL + 2*WT_PADL + H_PADL;
  int tid = threadIdx.x, lane = tid & 31, wave = tid >> 5;
  unsigned j = blockIdx.x, p = blockIdx.y;
  int i0 = p * (NUM_SLOTS / IPART), i1 = i0 + NUM_SLOTS / IPART;
  v8f acc[2] = {Z8, Z8};
  int icur = i0; if (icur == (int)j) icur++;
#ifdef HAVE_TDM
  if (icur < i1 && wave == 0) {                  // prologue DMA triple
    tdm_load_tile_f32((unsigned)(uintptr_t)stWs0,
        Wsrc + (unsigned)(icur*NUM_SLOTS + (int)j)*(D_MODEL*RANK), 256, 32, 32, 4, 1);
    tdm_load_tile_f32((unsigned)(uintptr_t)stWt0,
        Wtgt + (unsigned)(icur*NUM_SLOTS + (int)j)*(RANK*D_MODEL), 32, 256, 256, 7, 1);
    tdm_load_tile_f32((unsigned)(uintptr_t)stH0,
        hin + (unsigned)icur*D_MODEL, 8, 256, NUM_SLOTS*D_MODEL, 7, 1);
  }
#endif
  int bi = 0;
  while (icur < i1) {
    float* sWs = bi ? stWs1 : stWs0;
    float* sWt = bi ? stWt1 : stWt0;
    float* sHs = bi ? stH1  : stH0;
    float* nWs = bi ? stWs0 : stWs1;
    float* nWt = bi ? stWt0 : stWt1;
    float* nHs = bi ? stH0  : stH1;
    int inext = icur + 1; if (inext == (int)j) inext++;
    bool has_next = inext < i1;
#ifdef HAVE_TDM
    if (wave == 0 && has_next) {                 // DMA next triple into other buffers
      tdm_load_tile_f32((unsigned)(uintptr_t)nWs,
          Wsrc + (unsigned)(inext*NUM_SLOTS + (int)j)*(D_MODEL*RANK), 256, 32, 32, 4, 1);
      tdm_load_tile_f32((unsigned)(uintptr_t)nWt,
          Wtgt + (unsigned)(inext*NUM_SLOTS + (int)j)*(RANK*D_MODEL), 32, 256, 256, 7, 1);
      tdm_load_tile_f32((unsigned)(uintptr_t)nHs,
          hin + (unsigned)inext*D_MODEL, 8, 256, NUM_SLOTS*D_MODEL, 7, 1);
    }
    if (wave == 0) {
      if (has_next) __builtin_amdgcn_s_wait_tensorcnt(3);   // current triple done
      else          __builtin_amdgcn_s_wait_tensorcnt(0);
    }
#else
    {   // fallback: cooperative load of current tiles into padded staging
      const float* ws = Wsrc + (unsigned)(icur*NUM_SLOTS + (int)j)*(D_MODEL*RANK);
      const float* wt = Wtgt + (unsigned)(icur*NUM_SLOTS + (int)j)*(RANK*D_MODEL);
#pragma unroll
      for (int it = 0; it < 8; ++it) {
        unsigned idx = tid + it*256u, r = idx >> 3, c = (idx & 7)*4;
        float4 v = ((const float4*)ws)[idx];
        float* dst = sWs + r*33u + c;
        dst[0] = v.x; dst[1] = v.y; dst[2] = v.z; dst[3] = v.w;
      }
#pragma unroll
      for (int it = 0; it < 8; ++it) {
        unsigned idx = tid + it*256u, r = idx >> 6, c = (idx & 63)*4;
        float4 v = ((const float4*)wt)[idx];
        float* dst = sWt + r*257u + c;
        dst[0] = v.x; dst[1] = v.y; dst[2] = v.z; dst[3] = v.w;
      }
      for (int it = 0; it < 2; ++it) {
        unsigned idx = tid + it*256u, r = idx >> 6, c = (idx & 63)*4;
        float4 v = *(const float4*)(hin + ((r*NUM_SLOTS + (unsigned)icur)*D_MODEL + c));
        float* dst = sHs + r*257u + c;
        dst[0] = v.x; dst[1] = v.y; dst[2] = v.z; dst[3] = v.w;
      }
    }
#endif
    __syncthreads();
    if (wave < 2) {                              // GEMM1: P(16x32) = h @ Ws
      v8f pr = Z8;
#pragma unroll
      for (int k0 = 0; k0 < D_MODEL; k0 += 32) {
        v16bf af  = load_a_frag_f32z(sHs + k0, 257, lane);
        v16bf bfr = load_b_frag_kn_f32(sWs + (unsigned)k0*33u + wave*16u, 33, lane);
        pr = wmma_bf16(af, bfr, pr);
      }
      int n = lane & 15, mb = (lane >> 4) * 8;
#pragma unroll
      for (int r = 0; r < 8; ++r) sP[mb + r][wave*16 + n] = (__bf16)pr[r];
    }
    __syncthreads();
    // GEMM2: acc(16x32 cols per wave) += P(16x32) @ Wt(32x256)
    v16bf a2 = load_a_frag(&sP[0][0], 36, lane);
#pragma unroll
    for (int tn = 0; tn < 2; ++tn) {
      v16bf bfr = load_b_frag_kn_f32(sWt + wave*32u + tn*16u, 257, lane);
      acc[tn] = wmma_bf16(a2, bfr, acc[tn]);
    }
    __syncthreads();
    icur = inext; bi ^= 1;
  }
  // store partial tile rows 0..7 (batches); lanes 16..31 hold rows 8..15 (padding)
  unsigned n = lane & 15, mb = (lane >> 4) * 8;
  if (mb == 0) {
#pragma unroll
    for (int tn = 0; tn < 2; ++tn)
#pragma unroll
      for (int r = 0; r < 8; ++r)
        Pbuf[((p*NUM_SLOTS + j)*BATCH + r)*D_MODEL + wave*32u + tn*16u + n] = acc[tn][r];
  }
}

// ------------------------- combine partials + residual + ReLU + LayerNorm
__global__ __launch_bounds__(256) void combine_ln_kernel(
    const float* __restrict__ Pbuf, const float* __restrict__ hin,
    const float* __restrict__ gamma, const float* __restrict__ beta,
    float* __restrict__ hout)
{
  unsigned j = blockIdx.x;
  unsigned lane = threadIdx.x & 31, b = threadIdx.x >> 5;   // 8 waves == BATCH
  float x[8]; float mean = 0.f;
#pragma unroll
  for (int e = 0; e < 8; ++e) {
    unsigned d = lane + e*32;
    float infl = 0.f;
#pragma unroll
    for (int p = 0; p < IPART; ++p)
      infl += Pbuf[(((unsigned)p*NUM_SLOTS + j)*BATCH + b)*D_MODEL + d];
    float v = hin[(b*NUM_SLOTS + j)*D_MODEL + d] + fmaxf(infl, 0.f);
    x[e] = v; mean += v;
  }
#pragma unroll
  for (int o = 16; o > 0; o >>= 1) mean += __shfl_xor(mean, o);
  mean *= (1.f/256.f);
  float var = 0.f;
#pragma unroll
  for (int e = 0; e < 8; ++e) { float dl = x[e] - mean; var += dl*dl; }
#pragma unroll
  for (int o = 16; o > 0; o >>= 1) var += __shfl_xor(var, o);
  var *= (1.f/256.f);
  float rstd = rsqrtf(var + 1e-5f);
#pragma unroll
  for (int e = 0; e < 8; ++e) {
    unsigned d = lane + e*32;
    hout[(b*NUM_SLOTS + j)*D_MODEL + d] = (x[e] - mean) * rstd * gamma[d] + beta[d];
  }
}

// ---------------------------------- expand: fused scores->softmax->A2@Vf -> Y(B,L,D)
__global__ __launch_bounds__(256) void expand_kernel(
    const float* __restrict__ Q, const float* __restrict__ Kf,
    const float* __restrict__ Vf, float* __restrict__ Y)
{
  __shared__ __bf16 sQ[32][36];
  __shared__ __bf16 sK[128][36];
  __shared__ float  sS[32][129];
  __shared__ __bf16 sA2[32][132];
  __shared__ __bf16 sV[32][260];
  int tid = threadIdx.x, lane = tid & 31, wave = tid >> 5;
  unsigned b = blockIdx.y, l0 = blockIdx.x * 32u;
  v8f acc1[2] = {Z8, Z8};
  for (unsigned k0 = 0; k0 < D_MODEL; k0 += 32) {
    float4 rq, rk[4];
    {
      unsigned r = tid >> 3, c4 = tid & 7;       // 32x32 Q tile: 256 float4
      rq = *(const float4*)(Q + ((b*SEQ + l0 + r)*D_MODEL + k0 + c4*4u));
    }
#pragma unroll
    for (int it = 0; it < 4; ++it) {             // 128x32 K tile
      unsigned idx = tid + it*256u, r = idx >> 3, c4 = idx & 7;
      rk[it] = *(const float4*)(Kf + ((b*NUM_SLOTS + r)*D_MODEL + k0 + c4*4u));
    }
    {
      unsigned r = tid >> 3, c4 = tid & 7;
      *(bf16x4*)&sQ[r][c4*4] = cvt4(rq);
    }
#pragma unroll
    for (int it = 0; it < 4; ++it) {
      unsigned idx = tid + it*256u, r = idx >> 3, c4 = idx & 7;
      *(bf16x4*)&sK[r][c4*4] = cvt4(rk[it]);
    }
    __syncthreads();
    v16bf bfr = load_b_frag_nk(&sK[wave*16][0], 36, lane);
#pragma unroll
    for (int t = 0; t < 2; ++t) {
      v16bf af = load_a_frag(&sQ[t*16][0], 36, lane);
      acc1[t] = wmma_bf16(af, bfr, acc1[t]);
    }
    __syncthreads();
  }
  {
    int n = lane & 15, mb = (lane >> 4) * 8;
#pragma unroll
    for (int t = 0; t < 2; ++t)
#pragma unroll
      for (int r = 0; r < 8; ++r)
        sS[t*16 + mb + r][wave*16 + n] = acc1[t][r] * 0.0625f;
  }
  __syncthreads();
#pragma unroll
  for (int rr = 0; rr < 4; ++rr) {               // softmax over 128 slots
    int row = wave*4 + rr;
    float v[4]; float mx = -1e30f;
#pragma unroll
    for (int e = 0; e < 4; ++e) { v[e] = sS[row][lane*4 + e]; mx = fmaxf(mx, v[e]); }
#pragma unroll
    for (int o = 16; o > 0; o >>= 1) mx = fmaxf(mx, __shfl_xor(mx, o));
    float sum = 0.f;
#pragma unroll
    for (int e = 0; e < 4; ++e) { v[e] = __expf(v[e] - mx); sum += v[e]; }
#pragma unroll
    for (int o = 16; o > 0; o >>= 1) sum += __shfl_xor(sum, o);
    float inv = 1.f / sum;
    bf16x4 o4;
    o4[0] = (__bf16)(v[0]*inv); o4[1] = (__bf16)(v[1]*inv);
    o4[2] = (__bf16)(v[2]*inv); o4[3] = (__bf16)(v[3]*inv);
    *(bf16x4*)&sA2[row][lane*4] = o4;
  }
  __syncthreads();
  v8f acc2[2][2] = {{Z8, Z8}, {Z8, Z8}};
  for (unsigned kk0 = 0; kk0 < NUM_SLOTS; kk0 += 32) {
    float4 rv[8];
#pragma unroll
    for (int it = 0; it < 8; ++it) {             // Vf chunk 32x256: 2048 float4
      unsigned idx = tid + it*256u, r = idx >> 6, c4 = idx & 63;
      rv[it] = *(const float4*)(Vf + ((b*NUM_SLOTS + kk0 + r)*D_MODEL + c4*4u));
    }
#pragma unroll
    for (int it = 0; it < 8; ++it) {
      unsigned idx = tid + it*256u, r = idx >> 6, c4 = idx & 63;
      *(bf16x4*)&sV[r][c4*4] = cvt4(rv[it]);
    }
    __syncthreads();
#pragma unroll
    for (int mt = 0; mt < 2; ++mt) {
      v16bf af = load_a_frag(&sA2[mt*16][kk0], 132, lane);
#pragma unroll
      for (int tn = 0; tn < 2; ++tn) {
        v16bf bfr = load_b_frag_kn(&sV[0][wave*32 + tn*16], 260, lane);
        acc2[mt][tn] = wmma_bf16(af, bfr, acc2[mt][tn]);
      }
    }
    __syncthreads();
  }
  {
    unsigned n = lane & 15, mb = (lane >> 4) * 8;
#pragma unroll
    for (int mt = 0; mt < 2; ++mt)
#pragma unroll
      for (int tn = 0; tn < 2; ++tn)
#pragma unroll
        for (int r = 0; r < 8; ++r)
          Y[((b*SEQ + l0 + mt*16u + mb + r)*D_MODEL + wave*32u + tn*16u + n)] =
              acc2[mt][tn][r];
  }
}

// ---------------------------------------------------------------------------
extern "C" void kernel_launch(void* const* d_in, const int* in_sizes, int n_in,
                              void* d_out, int out_size, void* d_ws, size_t ws_size,
                              hipStream_t stream) {
  (void)in_sizes; (void)n_in; (void)out_size; (void)ws_size;
  const int*   ids    = (const int*)  d_in[0];
  const int*   amask  = (const int*)  d_in[1];
  const float* tok    = (const float*)d_in[2];
  const float* pos    = (const float*)d_in[3];
  const float* H      = (const float*)d_in[4];
  const float* Wsrc   = (const float*)d_in[5];
  const float* Wtgt   = (const float*)d_in[6];
  const float* Wq_in  = (const float*)d_in[7];
  const float* Wk_sl  = (const float*)d_in[8];
  const float* Wv_in  = (const float*)d_in[9];
  const float* Wq_out = (const float*)d_in[10];
  const float* Wk_f   = (const float*)d_in[11];
  const float* Wv_f   = (const float*)d_in[12];
  const float* Wout   = (const float*)d_in[13];
  const float* lng    = (const float*)d_in[14];
  const float* lnb    = (const float*)d_in[15];
  float* logits = (float*)d_out;

  char* ws = (char*)d_ws;
  size_t off = 0;
  auto take = [&](size_t nelts) -> float* {
    float* p = (float*)(ws + off);
    off += (nelts * 4 + 255) & ~(size_t)255;
    return p;
  };
  float* X    = take((size_t)BL*D_MODEL);
  float* Qin  = take((size_t)BL*D_MODEL);
  float* Vin  = take((size_t)BL*D_MODEL);
  float* Qout = take((size_t)BL*D_MODEL);
  float* Ksl  = take((size_t)NUM_SLOTS*D_MODEL);
  float* A    = take((size_t)BATCH*SEQ*NUM_SLOTS);
  float* csum = take((size_t)BATCH*NUM_SLOTS);
  float* h0   = take((size_t)BATCH*NUM_SLOTS*D_MODEL);
  float* h1   = take((size_t)BATCH*NUM_SLOTS*D_MODEL);
  float* Kf   = take((size_t)BATCH*NUM_SLOTS*D_MODEL);
  float* Vf   = take((size_t)BATCH*NUM_SLOTS*D_MODEL);
  float* Y    = take((size_t)BL*D_MODEL);
  float* Pbuf = take((size_t)IPART*NUM_SLOTS*BATCH*D_MODEL);

  dim3 blk(256);
  embed_kernel<<<dim3(BL*D_MODEL/1024), blk, 0, stream>>>(ids, tok, pos, X);

  gemm_awt<<<dim3(2, 64), blk, 0, stream>>>(X, Wq_in,  Qin,  BL, D_MODEL, D_MODEL);
  gemm_awt<<<dim3(2, 64), blk, 0, stream>>>(X, Wv_in,  Vin,  BL, D_MODEL, D_MODEL);
  gemm_awt<<<dim3(2, 64), blk, 0, stream>>>(X, Wq_out, Qout, BL, D_MODEL, D_MODEL);
  gemm_awt<<<dim3(2, 2),  blk, 0, stream>>>(H, Wk_sl,  Ksl,  NUM_SLOTS, D_MODEL, D_MODEL);

  compress_scores_kernel<<<dim3(SEQ/64, BATCH), blk, 0, stream>>>(Qin, Ksl, amask, A);
  colsum_kernel<<<dim3(BATCH*NUM_SLOTS/256), blk, 0, stream>>>(A, csum);
  compress_ir_kernel<<<dim3(2, 2, BATCH), blk, 0, stream>>>(A, csum, Vin, H, h0);

  size_t dynsz = (size_t)(2*WS_PADL + 2*WT_PADL + 2*H_PADL) * sizeof(float);  // ~146 KB
  float* hc = h0; float* hn = h1;
  for (int st = 0; st < STEPS; ++st) {
    step_partial_kernel<<<dim3(NUM_SLOTS, IPART), blk, dynsz, stream>>>(hc, Wsrc, Wtgt, Pbuf);
    combine_ln_kernel<<<dim3(NUM_SLOTS), blk, 0, stream>>>(
        Pbuf, hc, lng + st*D_MODEL, lnb + st*D_MODEL, hn);
    float* t = hc; hc = hn; hn = t;
  }

  gemm_awt<<<dim3(2, 16), blk, 0, stream>>>(hc, Wk_f, Kf, BATCH*NUM_SLOTS, D_MODEL, D_MODEL);
  gemm_awt<<<dim3(2, 16), blk, 0, stream>>>(hc, Wv_f, Vf, BATCH*NUM_SLOTS, D_MODEL, D_MODEL);

  expand_kernel<<<dim3(SEQ/32, BATCH), blk, 0, stream>>>(Qout, Kf, Vf, Y);

  gemm_awt<<<dim3(VOCAB/128, BL/64), blk, 0, stream>>>(Y, Wout, logits, BL, VOCAB, D_MODEL);
}
